// TransformerBlock_48241072668897
// MI455X (gfx1250) — matrix operand order repaired and checked by reading, not compiled, not run on hardware
//
#include <hip/hip_runtime.h>
#include <hip/hip_bf16.h>
#include <math.h>

// ---------------- problem constants ----------------
#define BB  4
#define LLN 1024
#define DDIM 1024
#define HHEADS 8
#define PPATH 1024
#define FFDIM 4096
#define DKK 128   // DDIM / HHEADS

typedef __attribute__((ext_vector_type(16))) _Float16 v16h;
typedef __attribute__((ext_vector_type(8)))  float    v8f;

// =====================================================================
// LayerNorm: one block (256 threads) per row of length D
// =====================================================================
__global__ __launch_bounds__(256)
void ln_kernel(const float* __restrict__ x, const float* __restrict__ g,
               const float* __restrict__ b, float* __restrict__ y, int D)
{
    const int row = blockIdx.x;
    const int tid = threadIdx.x;
    const float* xr = x + (size_t)row * D;
    float*       yr = y + (size_t)row * D;

    float s = 0.f, sq = 0.f;
    for (int i = tid; i < D; i += 256) {
        float v = xr[i];
        s += v; sq += v * v;
    }
    __shared__ float rs[256], rq[256];
    rs[tid] = s; rq[tid] = sq;
    __syncthreads();
    for (int off = 128; off > 0; off >>= 1) {
        if (tid < off) { rs[tid] += rs[tid + off]; rq[tid] += rq[tid + off]; }
        __syncthreads();
    }
    const float mean = rs[0] / D;
    const float var  = rq[0] / D - mean * mean;
    const float inv  = rsqrtf(var + 1e-6f);
    for (int i = tid; i < D; i += 256)
        yr[i] = g[i] * (xr[i] - mean) * inv + b[i];
}

// =====================================================================
// Generic WMMA GEMM:  C = act( A @ op(B) + bias ) (+ C if accum) (+ resid)
//   A: f32 row-major [M,K] (lda); B: f32 row-major [K,N] (ldb), or [N,K] if transB
//   C: f32 row-major [M,N] (ldc).  Two-level batch: z -> (b = z/Hdiv, h = z%Hdiv)
//
//   LDS tiles are stored FRAGMENT-MAJOR: each of the 128 threads stages the
//   exact contiguous 16-half (32B) slice that one (wave,lane) consumes, so
//   fragment reads are two ds_load_b128 each (no scalar ds_load_u16 packing).
//   f32 -> f16 conversion happens during staging; accumulation stays f32 via
//   v_wmma_f32_16x16x32_f16.  Block = 128 threads = 4 waves, 64x64 tile.
// =====================================================================
__global__ __launch_bounds__(128)
void gemm_wmma(const float* __restrict__ A, const float* __restrict__ Bm,
               float* __restrict__ C, const float* __restrict__ bias,
               const float* __restrict__ resid,
               int M, int N, int K, int lda, int ldb, int ldc,
               long long sAb, long long sAh, long long sBb, long long sBh,
               long long sCb, long long sCh,
               int Hdiv, int transB, int accum, int act)
{
    const int z  = blockIdx.z;
    const int bb = z / Hdiv;
    const int hh = z % Hdiv;
    A  += (size_t)bb * sAb + (size_t)hh * sAh;
    Bm += (size_t)bb * sBb + (size_t)hh * sBh;
    C  += (size_t)bb * sCb + (size_t)hh * sCh;
    const float* R = resid ? (resid + (size_t)bb * sCb + (size_t)hh * sCh) : nullptr;

    const int n0 = blockIdx.x * 64;
    const int m0 = blockIdx.y * 64;

    // fragment-major LDS: Af[wave][lane] and Bf[nt][lane], 16 halfs (32B) each
    __shared__ v16h Af[128];
    __shared__ v16h Bf[128];

    const int tid  = threadIdx.x;
    const int lane = tid & 31;          // wave32
    const int wv   = tid >> 5;          // wave id / staging group id (0..3)
    const int hf   = lane >> 4;         // lane half (0/1)
    const int fr   = lane & 15;         // row (A) / col (B) within 16x16 fragment

    // staging geometry (thread tid stages fragment slice (wv, lane))
    const int arow = m0 + wv * 16 + fr;          // A row this thread stages
    const int bcol = n0 + wv * 16 + fr;          // B col this thread stages (wv == nt)

    v8f acc[4];
    if (accum) {
        #pragma unroll
        for (int nt = 0; nt < 4; ++nt)
            #pragma unroll
            for (int r = 0; r < 8; ++r) {
                int row = m0 + wv * 16 + r + hf * 8;
                int col = n0 + nt * 16 + fr;
                acc[nt][r] = C[(size_t)row * ldc + col];
            }
    } else {
        #pragma unroll
        for (int nt = 0; nt < 4; ++nt)
            #pragma unroll
            for (int r = 0; r < 8; ++r) acc[nt][r] = 0.f;
    }

    for (int k0 = 0; k0 < K; k0 += 32) {
        // ---- stage A fragment slice: two contiguous 8-float runs ----
        {
            const float* ar = A + (size_t)arow * lda + k0 + hf * 16;
            v16h tmp;
            #pragma unroll
            for (int t = 0; t < 8; ++t)  tmp[t] = (_Float16)ar[t];        // K = t + 16*hf
            #pragma unroll
            for (int t = 8; t < 16; ++t) tmp[t] = (_Float16)ar[t];        // K = t + 16*hf
            Af[tid] = tmp;
        }
        // ---- stage B fragment slice: K = t + 16*hf for this lane ----
        {
            v16h tmp;
            if (!transB) {
                const float* br = Bm + (size_t)(k0 + 16 * hf) * ldb + bcol;
                #pragma unroll
                for (int t = 0; t < 16; ++t) tmp[t] = (_Float16)br[(size_t)t * ldb];
            } else {
                const float* br = Bm + (size_t)bcol * ldb + k0 + 16 * hf;
                #pragma unroll
                for (int t = 0; t < 16; ++t) tmp[t] = (_Float16)br[t];
            }
            Bf[tid] = tmp;
        }
        // CDNA5 prefetch of next A K-tile (global_prefetch_b8)
        if (k0 + 32 < K)
            __builtin_prefetch(&A[(size_t)arow * lda + (k0 + 32 + hf * 8)], 0, 1);
        __syncthreads();

        // ---- fragments: contiguous 32B loads (ds_load_b128 x2 each) ----
        v16h afrag = Af[wv * 32 + lane];
        #pragma unroll
        for (int nt = 0; nt < 4; ++nt) {
            v16h bfrag = Bf[nt * 32 + lane];
            acc[nt] = __builtin_amdgcn_wmma_f32_16x16x32_f16(
                false, afrag, false, bfrag, (short)0, acc[nt], false, false);
        }
        __syncthreads();
    }

    // ---- epilogue: bias, activation, residual, store ----
    #pragma unroll
    for (int nt = 0; nt < 4; ++nt) {
        #pragma unroll
        for (int r = 0; r < 8; ++r) {
            int row = m0 + wv * 16 + r + hf * 8;
            int col = n0 + nt * 16 + fr;
            float v = acc[nt][r];
            if (bias) v += bias[col];
            if (act) { // tanh-approx GELU (jax.nn.gelu default)
                float u = 0.7978845608028654f * (v + 0.044715f * v * v * v);
                v = 0.5f * v * (1.0f + tanhf(u));
            }
            if (R) v += R[(size_t)row * ldc + col];
            C[(size_t)row * ldc + col] = v;
        }
    }
}

// =====================================================================
// Fused score finalize + softmax:
//   S[b,h,i,j] = softmax_j( (S + QR[b,h,i,pm[b,i,j]]) / sqrt(DK) + ap, mask )
// One block (256 threads) per (b,h,i) row of length L=1024.
// =====================================================================
__global__ __launch_bounds__(256)
void attn_softmax(float* __restrict__ S, const float* __restrict__ QR,
                  const int* __restrict__ pm, const float* __restrict__ ap,
                  const int* __restrict__ mask)
{
    const int i = blockIdx.x, h = blockIdx.y, b = blockIdx.z;
    const int tid = threadIdx.x;
    const size_t srow = (((size_t)b * HHEADS + h) * LLN + i) * LLN;
    const float* qrr = QR + (((size_t)b * HHEADS + h) * LLN + i) * PPATH;
    const int*   pmr = pm + ((size_t)b * LLN + i) * LLN;
    const int*   mkr = mask + ((size_t)b * LLN + i) * LLN;   // mask is [B,1,L,L]
    const float* apr = ap + srow;
    const float scale = 0.08838834764831845f;                // 1/sqrt(128)

    float vals[4];
    float mx = -1e30f;
    #pragma unroll
    for (int c = 0; c < 4; ++c) {
        int j = tid + c * 256;
        float v = (S[srow + j] + qrr[pmr[j]]) * scale + apr[j];
        if (mkr[j] == 0) v = -1e9f;
        vals[c] = v;
        mx = fmaxf(mx, v);
    }
    __shared__ float red[256];
    red[tid] = mx; __syncthreads();
    for (int off = 128; off > 0; off >>= 1) {
        if (tid < off) red[tid] = fmaxf(red[tid], red[tid + off]);
        __syncthreads();
    }
    mx = red[0]; __syncthreads();

    float ssum = 0.f;
    #pragma unroll
    for (int c = 0; c < 4; ++c) {
        vals[c] = __expf(vals[c] - mx);
        ssum += vals[c];
    }
    red[tid] = ssum; __syncthreads();
    for (int off = 128; off > 0; off >>= 1) {
        if (tid < off) red[tid] += red[tid + off];
        __syncthreads();
    }
    const float inv = 1.0f / red[0];
    #pragma unroll
    for (int c = 0; c < 4; ++c)
        S[srow + tid + c * 256] = vals[c] * inv;
}

// =====================================================================
// Segment-sum scatter: AP[b,h,i,p] = sum_j attn[b,h,i,j] * (pm[b,i,j]==p)
// Per-row histogram in LDS (P=1024 floats = 4KB) via ds atomics.
// =====================================================================
__global__ __launch_bounds__(256)
void path_scatter(const float* __restrict__ S, const int* __restrict__ pm,
                  float* __restrict__ AP)
{
    const int i = blockIdx.x, h = blockIdx.y, b = blockIdx.z;
    const int tid = threadIdx.x;
    __shared__ float sp[PPATH];
    for (int c = tid; c < PPATH; c += 256) sp[c] = 0.f;
    __syncthreads();

    const size_t srow = (((size_t)b * HHEADS + h) * LLN + i) * LLN;
    const int* pmr = pm + ((size_t)b * LLN + i) * LLN;
    #pragma unroll
    for (int c = 0; c < 4; ++c) {
        int j = tid + c * 256;
        atomicAdd(&sp[pmr[j]], S[srow + j]);
    }
    __syncthreads();
    float* out = AP + (((size_t)b * HHEADS + h) * LLN + i) * PPATH;
    for (int c = tid; c < PPATH; c += 256) out[c] = sp[c];
}

// =====================================================================
// host-side launch helper
// =====================================================================
static void launch_gemm(hipStream_t s, const float* A, const float* B, float* C,
                        const float* bias, const float* resid,
                        int M, int N, int K, int lda, int ldb, int ldc,
                        long long sAb, long long sAh, long long sBb, long long sBh,
                        long long sCb, long long sCh,
                        int batch, int Hdiv, int transB, int accum, int act)
{
    dim3 g(N / 64, M / 64, batch), blk(128);
    hipLaunchKernelGGL(gemm_wmma, g, blk, 0, s, A, B, C, bias, resid,
                       M, N, K, lda, ldb, ldc,
                       sAb, sAh, sBb, sBh, sCb, sCh, Hdiv, transB, accum, act);
}

extern "C" void kernel_launch(void* const* d_in, const int* in_sizes, int n_in,
                              void* d_out, int out_size, void* d_ws, size_t ws_size,
                              hipStream_t stream)
{
    (void)in_sizes; (void)n_in; (void)out_size; (void)ws_size;

    const float* content = (const float*)d_in[0];
    const int*   mask    = (const int*)  d_in[1];
    const float* r_k     = (const float*)d_in[2];
    const float* r_v     = (const float*)d_in[3];
    const int*   pm      = (const int*)  d_in[4];
    const float* ap      = (const float*)d_in[5];
    const float* wq      = (const float*)d_in[6];
    const float* bq      = (const float*)d_in[7];
    const float* wk      = (const float*)d_in[8];
    const float* bk      = (const float*)d_in[9];
    const float* wv      = (const float*)d_in[10];
    const float* bv      = (const float*)d_in[11];
    const float* wo      = (const float*)d_in[12];
    const float* bo      = (const float*)d_in[13];
    const float* ln1_g   = (const float*)d_in[14];
    const float* ln1_b   = (const float*)d_in[15];
    const float* ln2_g   = (const float*)d_in[16];
    const float* ln2_b   = (const float*)d_in[17];
    const float* w1      = (const float*)d_in[18];
    const float* b1      = (const float*)d_in[19];
    const float* w2      = (const float*)d_in[20];
    const float* b2      = (const float*)d_in[21];
    float* out = (float*)d_out;

    // ---------------- workspace layout (floats) ----------------
    float* ws = (float*)d_ws;
    const size_t M1 = (size_t)1 << 20;
    float* xln  = ws + 0 * M1;    // [B*L, D]      4M
    float* Qlin = ws + 4 * M1;    // [B*L, D]      4M
    float* Klin = ws + 8 * M1;    // [B*L, D]      4M
    float* Vlin = ws + 12 * M1;   // [B*L, D]      4M
    float* ctx  = ws + 16 * M1;   // [B*L, D]      4M
    float* xres = ws + 20 * M1;   // [B*L, D]      4M
    float* xln2 = ws + 24 * M1;   // [B*L, D]      4M
    float* hbuf = ws + 28 * M1;   // [B*L, F]     16M
    float* S    = ws + 44 * M1;   // [B,H,L,L]    32M (scores -> attn)
    float* QR   = ws + 76 * M1;   // [B,H,L,P]    32M (qr -> attn_path)

    const int BL = BB * LLN;                        // 4096 rows
    const long long LD  = (long long)LLN * DDIM;    // per-b stride in Q/K/V/ctx
    const long long LL2 = (long long)LLN * LLN;     // per-h stride in S
    const long long LP  = (long long)LLN * PPATH;   // per-h stride in QR
    const long long PD  = (long long)PPATH * DKK;   // per-h stride in r_k/r_v

    // 1) LN1(content) -> xln
    hipLaunchKernelGGL(ln_kernel, dim3(BL), dim3(256), 0, stream,
                       content, ln1_g, ln1_b, xln, DDIM);

    // 2) Q/K/V projections: [4096,1024] @ [1024,1024] + bias
    launch_gemm(stream, xln, wq, Qlin, bq, nullptr, BL, DDIM, DDIM, DDIM, DDIM, DDIM,
                0, 0, 0, 0, 0, 0, 1, 1, 0, 0, 0);
    launch_gemm(stream, xln, wk, Klin, bk, nullptr, BL, DDIM, DDIM, DDIM, DDIM, DDIM,
                0, 0, 0, 0, 0, 0, 1, 1, 0, 0, 0);
    launch_gemm(stream, xln, wv, Vlin, bv, nullptr, BL, DDIM, DDIM, DDIM, DDIM, DDIM,
                0, 0, 0, 0, 0, 0, 1, 1, 0, 0, 0);

    // 3) scores S = Q @ K^T per (b,h):  [L,DK] x [L,DK]^T
    launch_gemm(stream, Qlin, Klin, S, nullptr, nullptr,
                LLN, LLN, DKK, DDIM, DDIM, LLN,
                LD, DKK, LD, DKK, (long long)HHEADS * LL2, LL2,
                BB * HHEADS, HHEADS, /*transB=*/1, 0, 0);

    // 4) QR = Q @ r_k^T per (b,h):  [L,DK] x [P,DK]^T
    launch_gemm(stream, Qlin, r_k, QR, nullptr, nullptr,
                LLN, PPATH, DKK, DDIM, DKK, PPATH,
                LD, DKK, (long long)HHEADS * PD, PD, (long long)HHEADS * LP, LP,
                BB * HHEADS, HHEADS, /*transB=*/1, 0, 0);

    // 5) gather + scale + bias + mask + softmax (in-place on S)
    hipLaunchKernelGGL(attn_softmax, dim3(LLN, HHEADS, BB), dim3(256), 0, stream,
                       S, QR, pm, ap, mask);

    // 6) attn_path = segment_sum(attn, pm) -> reuse QR buffer
    hipLaunchKernelGGL(path_scatter, dim3(LLN, HHEADS, BB), dim3(256), 0, stream,
                       S, pm, QR);

    // 7) ctx = attn @ V per (b,h), written straight into merged-head layout
    launch_gemm(stream, S, Vlin, ctx, nullptr, nullptr,
                LLN, DKK, LLN, LLN, DDIM, DDIM,
                (long long)HHEADS * LL2, LL2, LD, DKK, LD, DKK,
                BB * HHEADS, HHEADS, 0, 0, 0);

    // 8) ctx += attn_path @ r_v per (b,h)
    launch_gemm(stream, QR, r_v, ctx, nullptr, nullptr,
                LLN, DKK, PPATH, PPATH, DKK, DDIM,
                (long long)HHEADS * LP, LP, (long long)HHEADS * PD, PD, LD, DKK,
                BB * HHEADS, HHEADS, 0, /*accum=*/1, 0);

    // 9) x = ctx @ wo + bo + content   (fused residual)
    launch_gemm(stream, ctx, wo, xres, bo, content, BL, DDIM, DDIM, DDIM, DDIM, DDIM,
                0, 0, 0, 0, 0, 0, 1, 1, 0, 0, 0);

    // 10) LN2(x) -> xln2
    hipLaunchKernelGGL(ln_kernel, dim3(BL), dim3(256), 0, stream,
                       xres, ln2_g, ln2_b, xln2, DDIM);

    // 11) h = gelu(xln2 @ w1 + b1)
    launch_gemm(stream, xln2, w1, hbuf, b1, nullptr, BL, FFDIM, DDIM, DDIM, FFDIM, FFDIM,
                0, 0, 0, 0, 0, 0, 1, 1, 0, 0, /*gelu=*/1);

    // 12) out = h @ w2 + b2 + x   (fused residual, final output)
    launch_gemm(stream, hbuf, w2, out, b2, xres, BL, DDIM, FFDIM, FFDIM, DDIM, DDIM,
                0, 0, 0, 0, 0, 0, 1, 1, 0, 0, 0);
}
